// TritterAttention_27573690040805
// MI455X (gfx1250) — matrix operand order repaired and checked by reading, not compiled, not run on hardware
//
#include <hip/hip_runtime.h>

typedef __attribute__((ext_vector_type(16))) _Float16 v16h;
typedef __attribute__((ext_vector_type(8)))  float    v8f;
typedef __attribute__((ext_vector_type(4)))  int      v4i;

#define NEG_INF (-__builtin_inff())
#define AS1 __attribute__((address_space(1)))
#define AS3 __attribute__((address_space(3)))

#if __has_builtin(__builtin_amdgcn_global_load_async_to_lds_b128) && \
    __has_builtin(__builtin_amdgcn_s_wait_asynccnt)
#define HAVE_ASYNC_LDS 1
#else
#define HAVE_ASYNC_LDS 0
#endif

constexpr int S_LEN = 4096;
constexpr int HDIM  = 1024;
constexpr int NH_   = 16;
constexpr int HD_   = 64;
constexpr int LKA   = 40;   // LDS row stride (halves) for GEMM k-tiles (32 + 8 pad)
constexpr int LFA   = 72;   // LDS row stride (halves) for flash tiles   (64 + 8 pad)

// ---------------------------------------------------------------------------
// WMMA helpers (CDNA5 gfx1250, wave32). D = A(16x32 f16) * B(32x16 f16) + C(f32)
// ---------------------------------------------------------------------------
__device__ __forceinline__ v8f wmma_f16(v16h a, v16h b, v8f c) {
  // (neg_a, A, neg_b, B, c_mod, C, reuse_a, reuse_b)
  return __builtin_amdgcn_wmma_f32_16x16x32_f16(false, a, false, b, (short)0, c,
                                                false, false);
}

// A-matrix 16x32 f16 fragment. Lane layout (ISA 7.12.2):
//   lane l: M = l%16, hi = l/16;  a[j] = A[M][kk + (j/8)*16 + hi*8 + (j%8)]
// => two contiguous 8-half (16B aligned) runs per lane.
__device__ __forceinline__ v16h frag_a16(const _Float16* base, int stride,
                                         int row, int hi, int kk) {
  v16h a;
  const _Float16* p = base + row * stride + kk + hi * 8;
#pragma unroll
  for (int j = 0; j < 8; ++j) { a[j] = p[j]; a[j + 8] = p[16 + j]; }
  return a;
}

// B-matrix 32x16 f16 fragment from an LDS tile stored [n][k] (row n contiguous k):
//   lane l: N = l%16, hi = l/16;  b[j] = tile[N][kk + (j/8)*16 + hi*8 + (j%8)]
// => two contiguous 8-half (16B aligned) runs per lane.
__device__ __forceinline__ v16h frag_b16(const _Float16* base, int stride,
                                         int n, int hi, int kk) {
  v16h b;
  const _Float16* p = base + n * stride + kk + hi * 8;
#pragma unroll
  for (int j = 0; j < 8; ++j) { b[j] = p[j]; b[j + 8] = p[16 + j]; }
  return b;
}

// Async 16B global -> LDS copy (per lane), with synchronous fallback.
// Builtin prototype (from hipcc diagnostic): (v4i __device__*, v4i __shared__*,
// imm offset, imm cpol).
__device__ __forceinline__ void stage16(const _Float16* g, _Float16* l) {
#if HAVE_ASYNC_LDS
  __builtin_amdgcn_global_load_async_to_lds_b128((AS1 v4i*)(void*)g,
                                                 (AS3 v4i*)(void*)l, 0, 0);
#else
  *(uint4*)l = *(const uint4*)g;
#endif
}
__device__ __forceinline__ void stage_wait() {
#if HAVE_ASYNC_LDS
  __builtin_amdgcn_s_wait_asynccnt(0);
#endif
}

// ---------------------------------------------------------------------------
// 64x128 GEMM core: Y(m0:+64, n0:+128) = A(MxK) * W^T, W stored [N][K] f32.
// 128 threads = 4 waves; wave (wr,wc) owns a 32x64 subtile:
//   2 A-frags + 4 B-frags -> 8 WMMAs per 32-wide K step (B-frags reused twice).
// ---------------------------------------------------------------------------
template <bool AF16>
__device__ __forceinline__ void gemm_64x128(const void* __restrict__ Ap, int lda,
                                            const float* __restrict__ W, int Kdim,
                                            int m0, int n0,
                                            _Float16 (&Xs)[64][LKA],
                                            _Float16 (&Ws)[128][LKA],
                                            v8f (&acc)[2][4]) {
  const int tid  = threadIdx.x;
  const int wave = tid >> 5, lane = tid & 31;
  const int hi = lane >> 4, ln = lane & 15;
  const int wr = wave >> 1, wc = wave & 1;
  const int lrow = tid >> 1, lhalf = tid & 1;  // A tile: 64 rows x 2 halves

  for (int kt = 0; kt < Kdim; kt += 32) {
    __syncthreads();
    if constexpr (AF16) {
      stage16((const _Float16*)Ap + (size_t)(m0 + lrow) * lda + kt + lhalf * 16,
              &Xs[lrow][lhalf * 16]);
      stage16((const _Float16*)Ap + (size_t)(m0 + lrow) * lda + kt + lhalf * 16 + 8,
              &Xs[lrow][lhalf * 16 + 8]);
    } else {
      const float* a = (const float*)Ap + (size_t)(m0 + lrow) * lda + kt + lhalf * 16;
#pragma unroll
      for (int j = 0; j < 16; ++j) Xs[lrow][lhalf * 16 + j] = (_Float16)a[j];
    }
    {  // W tile 128x32: thread tid owns one full row
      const float* w = W + (size_t)(n0 + tid) * Kdim + kt;
#pragma unroll
      for (int j = 0; j < 32; ++j) Ws[tid][j] = (_Float16)w[j];
    }
    stage_wait();
    __syncthreads();

    const v16h a0 = frag_a16(&Xs[0][0], LKA, wr * 32 + ln,      hi, 0);
    const v16h a1 = frag_a16(&Xs[0][0], LKA, wr * 32 + 16 + ln, hi, 0);
#pragma unroll
    for (int ni = 0; ni < 4; ++ni) {
      const v16h b = frag_b16(&Ws[0][0], LKA, wc * 64 + ni * 16 + ln, hi, 0);
      acc[0][ni] = wmma_f16(a0, b, acc[0][ni]);
      acc[1][ni] = wmma_f16(a1, b, acc[1][ni]);
    }
  }
}

// ---------------------------------------------------------------------------
// Phase 1: Q/K/V = X @ W^T + b, written f16 into per-head [NH][S][HD] layout.
// grid = (S/64, H/128, 3), block = 128
// ---------------------------------------------------------------------------
__global__ __launch_bounds__(128) void qkv_proj_kernel(
    const float* __restrict__ X,
    const float* __restrict__ Wq, const float* __restrict__ Wk,
    const float* __restrict__ Wv,
    const float* __restrict__ bq, const float* __restrict__ bk,
    const float* __restrict__ bv,
    _Float16* __restrict__ Qh, _Float16* __restrict__ Kh,
    _Float16* __restrict__ Vh) {
  __shared__ __align__(16) _Float16 Xs[64][LKA];
  __shared__ __align__(16) _Float16 Ws[128][LKA];

  const int m0 = blockIdx.x * 64, n0 = blockIdx.y * 128, which = blockIdx.z;
  const float* W    = (which == 0) ? Wq : (which == 1) ? Wk : Wv;
  const float* bias = (which == 0) ? bq : (which == 1) ? bk : bv;
  _Float16* Out     = (which == 0) ? Qh : (which == 1) ? Kh : Vh;

  v8f acc[2][4];
#pragma unroll
  for (int i = 0; i < 2; ++i)
#pragma unroll
    for (int j = 0; j < 4; ++j)
#pragma unroll
      for (int e = 0; e < 8; ++e) acc[i][j][e] = 0.f;

  gemm_64x128<false>(X, HDIM, W, HDIM, m0, n0, Xs, Ws, acc);

  const int tid = threadIdx.x, wave = tid >> 5, lane = tid & 31;
  const int hi = lane >> 4, ln = lane & 15;
  const int wr = wave >> 1, wc = wave & 1;
#pragma unroll
  for (int mi = 0; mi < 2; ++mi)
#pragma unroll
    for (int ni = 0; ni < 4; ++ni) {
      const int n = n0 + wc * 64 + ni * 16 + ln;
      const float bb = bias[n];
      const int h = n >> 6, d = n & 63;
#pragma unroll
      for (int r = 0; r < 8; ++r) {
        const int m = m0 + wr * 32 + mi * 16 + r + 8 * hi;
        Out[(size_t)h * S_LEN * HD_ + (size_t)m * HD_ + d] =
            (_Float16)(acc[mi][ni][r] + bb);
      }
    }
}

// ---------------------------------------------------------------------------
// Phase 2: flash attention per head (causal mask applied analytically).
// grid = (S/64, NH), block = 128; each wave owns 16 Q rows with online softmax.
// ---------------------------------------------------------------------------
__global__ __launch_bounds__(128) void flash_attn_kernel(
    const _Float16* __restrict__ Q, const _Float16* __restrict__ K,
    const _Float16* __restrict__ V, _Float16* __restrict__ O) {
  __shared__ __align__(16) _Float16 Qs[64][LFA];
  __shared__ __align__(16) _Float16 Ks[64][LFA];
  __shared__ __align__(16) _Float16 Vs[64][LFA];  // transposed: Vs[d][j] = V[j][d]
  __shared__ __align__(16) _Float16 Ps[64][LFA];  // per-wave P staging (C->A layout)

  const int h = blockIdx.y, qt = blockIdx.x;
  const int tid = threadIdx.x, wave = tid >> 5, lane = tid & 31;
  const int hi = lane >> 4, ln = lane & 15;
  const int row = tid >> 1, half = tid & 1;
  const size_t hbase = (size_t)h * S_LEN * HD_;
  const float SCALE = 0.125f;  // 1/sqrt(64)

  {  // stage Q tile (64x64 f16) once, async where available
    const _Float16* g = Q + hbase + (size_t)(qt * 64 + row) * HD_ + half * 32;
    _Float16* l = &Qs[row][half * 32];
#pragma unroll
    for (int c = 0; c < 4; ++c) stage16(g + c * 8, l + c * 8);
  }

  float m_i[8], l_i[8];
  v8f accO[4];
#pragma unroll
  for (int r = 0; r < 8; ++r) { m_i[r] = NEG_INF; l_i[r] = 0.f; }
#pragma unroll
  for (int dt = 0; dt < 4; ++dt)
#pragma unroll
    for (int r = 0; r < 8; ++r) accO[dt][r] = 0.f;

  stage_wait();
  __syncthreads();
  const v16h aq0 = frag_a16(&Qs[0][0], LFA, wave * 16 + ln, hi, 0);
  const v16h aq1 = frag_a16(&Qs[0][0], LFA, wave * 16 + ln, hi, 32);

  for (int t = 0; t <= qt; ++t) {
    __syncthreads();
    {  // K row-major (async), V transposed (manual)
      const _Float16* gk = K + hbase + (size_t)(t * 64 + row) * HD_ + half * 32;
      _Float16* lk = &Ks[row][half * 32];
#pragma unroll
      for (int c = 0; c < 4; ++c) stage16(gk + c * 8, lk + c * 8);

      const _Float16* sv = V + hbase + (size_t)(t * 64 + row) * HD_ + half * 32;
      _Float16 tmp[32];
#pragma unroll
      for (int d0 = 0; d0 < 32; ++d0) tmp[d0] = sv[d0];
#pragma unroll
      for (int d0 = 0; d0 < 32; ++d0) Vs[half * 32 + d0][row] = tmp[d0];
    }
    if (t + 1 <= qt) {  // global_prefetch_b8 of next tiles
      __builtin_prefetch(K + hbase + (size_t)((t + 1) * 64 + row) * HD_ + half * 32, 0, 0);
      __builtin_prefetch(V + hbase + (size_t)((t + 1) * 64 + row) * HD_ + half * 32, 0, 0);
    }
    stage_wait();
    __syncthreads();

    // S = Q * K^T : 16x64 scores per wave, 4 col-tiles x 2 d-chunks
    v8f sc[4];
#pragma unroll
    for (int nt = 0; nt < 4; ++nt) {
#pragma unroll
      for (int e = 0; e < 8; ++e) sc[nt][e] = 0.f;
      const v16h bk0 = frag_b16(&Ks[0][0], LFA, nt * 16 + ln, hi, 0);
      sc[nt] = wmma_f16(aq0, bk0, sc[nt]);
      const v16h bk1 = frag_b16(&Ks[0][0], LFA, nt * 16 + ln, hi, 32);
      sc[nt] = wmma_f16(aq1, bk1, sc[nt]);
    }

    // scale + causal mask (diagonal tile only)
    if (t == qt) {
#pragma unroll
      for (int nt = 0; nt < 4; ++nt) {
        const int c = t * 64 + nt * 16 + ln;
#pragma unroll
        for (int r = 0; r < 8; ++r) {
          const int rg = qt * 64 + wave * 16 + r + 8 * hi;
          sc[nt][r] = (c <= rg) ? sc[nt][r] * SCALE : NEG_INF;
        }
      }
    } else {
#pragma unroll
      for (int nt = 0; nt < 4; ++nt)
#pragma unroll
        for (int r = 0; r < 8; ++r) sc[nt][r] *= SCALE;
    }

    // online softmax: row max/sum across the 16-lane half holding each row
    float alpha[8];
#pragma unroll
    for (int r = 0; r < 8; ++r) {
      float m = fmaxf(fmaxf(sc[0][r], sc[1][r]), fmaxf(sc[2][r], sc[3][r]));
#pragma unroll
      for (int off = 1; off < 16; off <<= 1) m = fmaxf(m, __shfl_xor(m, off, 32));
      const float mn = fmaxf(m_i[r], m);
      alpha[r] = __expf(m_i[r] - mn);
      m_i[r] = mn;
    }
#pragma unroll
    for (int nt = 0; nt < 4; ++nt)
#pragma unroll
      for (int r = 0; r < 8; ++r) sc[nt][r] = __expf(sc[nt][r] - m_i[r]);
#pragma unroll
    for (int r = 0; r < 8; ++r) {
      float s = (sc[0][r] + sc[1][r]) + (sc[2][r] + sc[3][r]);
#pragma unroll
      for (int off = 1; off < 16; off <<= 1) s += __shfl_xor(s, off, 32);
      l_i[r] = l_i[r] * alpha[r] + s;
    }
#pragma unroll
    for (int dt = 0; dt < 4; ++dt)
#pragma unroll
      for (int r = 0; r < 8; ++r) accO[dt][r] *= alpha[r];

    // P (C-layout) -> per-wave LDS region -> A-layout fragments (DS in-order
    // within a wave; no barrier needed for the wave-private region)
#pragma unroll
    for (int nt = 0; nt < 4; ++nt)
#pragma unroll
      for (int r = 0; r < 8; ++r)
        Ps[wave * 16 + r + 8 * hi][nt * 16 + ln] = (_Float16)sc[nt][r];
    const v16h ap0 = frag_a16(&Ps[0][0], LFA, wave * 16 + ln, hi, 0);
    const v16h ap1 = frag_a16(&Ps[0][0], LFA, wave * 16 + ln, hi, 32);

    // O += P * V  (B from transposed Vs => contiguous runs)
#pragma unroll
    for (int dt = 0; dt < 4; ++dt) {
      const v16h bv0 = frag_b16(&Vs[0][0], LFA, dt * 16 + ln, hi, 0);
      accO[dt] = wmma_f16(ap0, bv0, accO[dt]);
      const v16h bv1 = frag_b16(&Vs[0][0], LFA, dt * 16 + ln, hi, 32);
      accO[dt] = wmma_f16(ap1, bv1, accO[dt]);
    }
  }

  // finalize: divide by row sums, store f16 to [S][H] layout for the out-proj
#pragma unroll
  for (int r = 0; r < 8; ++r) l_i[r] = 1.f / l_i[r];
#pragma unroll
  for (int dt = 0; dt < 4; ++dt)
#pragma unroll
    for (int r = 0; r < 8; ++r) {
      const int m = qt * 64 + wave * 16 + r + 8 * hi;
      O[(size_t)m * HDIM + h * HD_ + dt * 16 + ln] =
          (_Float16)(accO[dt][r] * l_i[r]);
    }
}

// ---------------------------------------------------------------------------
// Phase 3: out = attn_out @ Wo^T + bo  (f16 A, f32 result)
// grid = (S/64, H/128), block = 128
// ---------------------------------------------------------------------------
__global__ __launch_bounds__(128) void out_proj_kernel(
    const _Float16* __restrict__ A, const float* __restrict__ Wo,
    const float* __restrict__ bo, float* __restrict__ Y) {
  __shared__ __align__(16) _Float16 Xs[64][LKA];
  __shared__ __align__(16) _Float16 Ws[128][LKA];

  const int m0 = blockIdx.x * 64, n0 = blockIdx.y * 128;
  v8f acc[2][4];
#pragma unroll
  for (int i = 0; i < 2; ++i)
#pragma unroll
    for (int j = 0; j < 4; ++j)
#pragma unroll
      for (int e = 0; e < 8; ++e) acc[i][j][e] = 0.f;

  gemm_64x128<true>(A, HDIM, Wo, HDIM, m0, n0, Xs, Ws, acc);

  const int tid = threadIdx.x, wave = tid >> 5, lane = tid & 31;
  const int hi = lane >> 4, ln = lane & 15;
  const int wr = wave >> 1, wc = wave & 1;
#pragma unroll
  for (int mi = 0; mi < 2; ++mi)
#pragma unroll
    for (int ni = 0; ni < 4; ++ni) {
      const int n = n0 + wc * 64 + ni * 16 + ln;
      const float bb = bo[n];
#pragma unroll
      for (int r = 0; r < 8; ++r) {
        const int m = m0 + wr * 32 + mi * 16 + r + 8 * hi;
        Y[(size_t)m * HDIM + n] = acc[mi][ni][r] + bb;
      }
    }
}

// ---------------------------------------------------------------------------
// Launch: QKV proj -> flash attention -> output proj (stream-ordered).
// Workspace: Q,K,V,attn_out as f16 = 4 * 8MB = 32MB.
// ---------------------------------------------------------------------------
extern "C" void kernel_launch(void* const* d_in, const int* in_sizes, int n_in,
                              void* d_out, int out_size, void* d_ws,
                              size_t ws_size, hipStream_t stream) {
  (void)in_sizes; (void)n_in; (void)out_size; (void)ws_size;
  const float* X  = (const float*)d_in[0];
  // d_in[1] = attention_mask: pure causal additive mask, applied analytically.
  const float* wq = (const float*)d_in[2];
  const float* bq = (const float*)d_in[3];
  const float* wk = (const float*)d_in[4];
  const float* bk = (const float*)d_in[5];
  const float* wv = (const float*)d_in[6];
  const float* bv = (const float*)d_in[7];
  const float* wo = (const float*)d_in[8];
  const float* bo = (const float*)d_in[9];
  float* out = (float*)d_out;

  const size_t elems = (size_t)NH_ * S_LEN * HD_;  // 4M halves = 8MB
  _Float16* Qh = (_Float16*)d_ws;
  _Float16* Kh = Qh + elems;
  _Float16* Vh = Kh + elems;
  _Float16* Oh = Vh + elems;

  qkv_proj_kernel<<<dim3(S_LEN / 64, HDIM / 128, 3), 128, 0, stream>>>(
      X, wq, wk, wv, bq, bk, bv, Qh, Kh, Vh);
  flash_attn_kernel<<<dim3(S_LEN / 64, NH_), 128, 0, stream>>>(Qh, Kh, Vh, Oh);
  out_proj_kernel<<<dim3(S_LEN / 64, HDIM / 128), 128, 0, stream>>>(Oh, wo, bo, out);
}
